// GATEncoder_35605278883997
// MI455X (gfx1250) — compile-verified, matrix operand order checked by reading, not run on hardware
//
#include <hip/hip_runtime.h>
#include <cstdint>
#include <cstddef>

// ---------------------------------------------------------------------------
// GAT 2-layer encoder for MI455X (gfx1250, wave32, WMMA).
//   L1: h1 = x @ W1            (WMMA f16->f32, K=512, COLS=64, BM=64)
//       alpha_s/d = <h1, att>  per (node, head)
//       softmax-agg over edges (exp without max-shift: mathematically equal)
//       out1 = ELU(agg/s + b1)
//   L2: h2 = out1 @ W2         (WMMA, K=64, COLS=128, BM=32)
//       1-head softmax-agg -> d_out, finalize adds b2.
// ---------------------------------------------------------------------------

typedef _Float16 half4  __attribute__((ext_vector_type(4)));
typedef _Float16 half8  __attribute__((ext_vector_type(8)));
typedef _Float16 half16 __attribute__((ext_vector_type(16)));
typedef float    f32x4  __attribute__((ext_vector_type(4)));
typedef float    f32x8  __attribute__((ext_vector_type(8)));

// ------------------------------ utility kernels ----------------------------

__global__ void fill_zero(float* __restrict__ p, long n) {
  long i = (long)blockIdx.x * blockDim.x + threadIdx.x;
  if (i < n) p[i] = 0.0f;
}

// out[c*K + k] = (f16) in[k*C + c]   (transpose + convert, for W matrices)
__global__ void convertT(const float* __restrict__ in, _Float16* __restrict__ out,
                         int K, int C) {
  int i = blockIdx.x * blockDim.x + threadIdx.x;
  if (i >= K * C) return;
  int c = i / K, k = i % K;
  out[(size_t)c * K + k] = (_Float16)in[(size_t)k * C + c];
}

// ------------------------------ WMMA GEMM ----------------------------------
// C[nrows, COLS] = A[nrows, K] (f32) x BT[COLS, K] (f16, pre-transposed).
// Block: 256 threads = 8 waves. Tile grid: (BM/16) x (COLS/16) 16x16 tiles;
// wave w owns col-tile nt = w % NT and MT m-tiles {mg + j*(8/NT)}, sharing one
// B fragment across its MT WMMAs per k-step. A staged via double-buffered LDS
// (one barrier per k-step). Out-of-range rows are clamped on load (their LDS
// rows feed only epilogue-discarded outputs), so the loader is branch-free.
template <int BM, int COLS, int K>
__global__ __launch_bounds__(256)
void gemm_wmma(const float* __restrict__ A, const _Float16* __restrict__ BT,
               float* __restrict__ C, int nrows) {
  constexpr int KSTEP = 32;
  constexpr int LDA   = KSTEP + 8;          // 40 halfs = 80B rows (16B aligned)
  __shared__ _Float16 As[2][BM][LDA];       // double-buffered A tile

  constexpr int NT      = COLS / 16;        // col tiles
  constexpr int MROWS   = BM / 16;          // m tiles
  constexpr int MT      = (MROWS * NT) / 8; // m tiles per wave
  constexpr int MSTRIDE = 8 / NT;
  static_assert(MT >= 1 && MROWS * NT == 8 * MT, "tile mapping");

  const int tid  = threadIdx.x;
  const int wv   = tid >> 5;
  const int lane = tid & 31;
  const int hlf  = lane >> 4;               // 0: lanes 0-15, 1: lanes 16-31
  const int lm   = lane & 15;
  const int nt   = wv % NT;
  const int mg   = wv / NT;
  const long rowB = (long)blockIdx.x * BM;

  // cooperative A-tile loader: branch-free (row clamped into range)
  constexpr int LPT = (BM * KSTEP) / 256;   // f32 elems per thread
  constexpr int TPR = KSTEP / LPT;          // threads per row
  static_assert(LPT % 4 == 0, "chunked loader");
  constexpr int NCH = LPT / 4;
  const int lr = tid / TPR;
  const int lk = (tid % TPR) * LPT;
  long rowA = rowB + lr;
  if (rowA >= nrows) rowA = nrows - 1;      // clamp: padding rows are discarded
  const float* arow = A + rowA * (long)K;

  f32x8 acc[MT];
#pragma unroll
  for (int j = 0; j < MT; ++j) acc[j] = (f32x8){};

  int buf = 0;
  for (int kb = 0; kb < K; kb += KSTEP, buf ^= 1) {
    if (kb + KSTEP < K)
      __builtin_prefetch(arow + kb + KSTEP + lk, 0, 3);   // global_prefetch_b8

    // issue all chunk loads first (clauses), then convert+store
    f32x4 va[NCH];
#pragma unroll
    for (int j = 0; j < NCH; ++j)
      va[j] = *(const f32x4*)(arow + kb + lk + j * 4);
#pragma unroll
    for (int j = 0; j < NCH; ++j) {
      half4 vh;
#pragma unroll
      for (int i = 0; i < 4; ++i) vh[i] = (_Float16)va[j][i];
      *(half4*)(&As[buf][lr][lk + j * 4]) = vh;           // one ds_store_b64
    }
    __syncthreads();   // single barrier: write(buf) -> read(buf); read(buf) vs
                       // write(buf) two iters later is separated transitively

    // B fragment (32x16): lane(hlf,n) holds K chunks hlf*8+[0..7], 16+hlf*8+[0..7]
    const _Float16* br = BT + (long)(nt * 16 + lm) * K + kb;
    half8 blo = *(const half8*)(br + hlf * 8);
    half8 bhi = *(const half8*)(br + 16 + hlf * 8);
    half16 bf;
#pragma unroll
    for (int i = 0; i < 8; ++i) { bf[i] = blo[i]; bf[i + 8] = bhi[i]; }

#pragma unroll
    for (int j = 0; j < MT; ++j) {
      const _Float16* ar = &As[buf][(mg + j * MSTRIDE) * 16 + lm][0];
      half8 alo = *(const half8*)(ar + hlf * 8);
      half8 ahi = *(const half8*)(ar + 16 + hlf * 8);
      half16 af;
#pragma unroll
      for (int i = 0; i < 8; ++i) { af[i] = alo[i]; af[i + 8] = ahi[i]; }

      acc[j] = __builtin_amdgcn_wmma_f32_16x16x32_f16(
          /*neg_a=*/false, af, /*neg_b=*/false, bf,
          /*c_mod=*/(short)0, acc[j], /*reuse_a=*/false, /*reuse_b=*/false);
    }
  }

  // D layout: VGPR r -> row = r + 8*hlf, col = lm
#pragma unroll
  for (int j = 0; j < MT; ++j) {
#pragma unroll
    for (int r = 0; r < 8; ++r) {
      long row = rowB + (mg + j * MSTRIDE) * 16 + r + 8 * hlf;
      if (row < nrows) C[row * COLS + nt * 16 + lm] = acc[j][r];
    }
  }
}

// ------------------------ attention score kernels --------------------------

// alpha_s[n,h] = sum_c h[n,h,c]*a_src[h,c];  alpha_d likewise.
__global__ void scores_k(const float* __restrict__ h, const float* __restrict__ a_src,
                         const float* __restrict__ a_dst, float* __restrict__ os,
                         float* __restrict__ od, int N, int H, int Cc) {
  int i = blockIdx.x * blockDim.x + threadIdx.x;
  if (i >= N * H) return;
  int n = i / H, hh = i % H;
  const float* hp = h + (long)n * H * Cc + (long)hh * Cc;
  float ss = 0.0f, sd = 0.0f;
  for (int c = 0; c < Cc; ++c) {
    float v = hp[c];
    ss += v * a_src[hh * Cc + c];
    sd += v * a_dst[hh * Cc + c];
  }
  os[i] = ss;
  od[i] = sd;
}

// ---------------------------- edge passes ----------------------------------
// Edges: e < E from edge_index (int64, [2,E]); e >= E are self-loops (e-E).
// Softmax without max-shift (identical result: scores are O(1)).

__device__ __forceinline__ float lrelu02(float v) {
  return v > 0.0f ? v : 0.2f * v;
}

template <int H>
__global__ void edge_sum(const long long* __restrict__ ei, int E, int N,
                         const float* __restrict__ as, const float* __restrict__ ad,
                         float* __restrict__ s) {
  long e = (long)blockIdx.x * blockDim.x + threadIdx.x;
  long ET = (long)E + N;
  if (e >= ET) return;
  int src, dst;
  if (e < E) { src = (int)ei[e]; dst = (int)ei[(long)E + e]; }
  else       { src = dst = (int)(e - E); }
#pragma unroll
  for (int hh = 0; hh < H; ++hh) {
    float v = lrelu02(as[(long)src * H + hh] + ad[(long)dst * H + hh]);
    atomicAdd(&s[(long)dst * H + hh], expf(v));
  }
}

template <int H, int Cc>
__global__ void edge_agg(const long long* __restrict__ ei, int E, int N,
                         const float* __restrict__ as, const float* __restrict__ ad,
                         const float* __restrict__ h, float* __restrict__ acc) {
  long e = (long)blockIdx.x * blockDim.x + threadIdx.x;
  long ET = (long)E + N;
  if (e >= ET) return;
  int src, dst;
  if (e < E) { src = (int)ei[e]; dst = (int)ei[(long)E + e]; }
  else       { src = dst = (int)(e - E); }

  const float* hp = h + (long)src * H * Cc;
  float* ap = acc + (long)dst * H * Cc;
  __builtin_prefetch(hp, 0, 3);

  float p[H];
#pragma unroll
  for (int hh = 0; hh < H; ++hh)
    p[hh] = expf(lrelu02(as[(long)src * H + hh] + ad[(long)dst * H + hh]));

#pragma unroll 8
  for (int f = 0; f < H * Cc; ++f)
    atomicAdd(&ap[f], p[f / Cc] * hp[f]);
}

// ---------------------------- finalize kernels -----------------------------

// out1 = ELU(acc/s + b1), in place (8 heads x 8 ch)
__global__ void finalize1(float* __restrict__ acc, const float* __restrict__ s,
                          const float* __restrict__ b, int N) {
  int i = blockIdx.x * blockDim.x + threadIdx.x;
  if (i >= N * 64) return;
  int n = i >> 6, j = i & 63, hh = j >> 3;
  float v = acc[i] / s[n * 8 + hh] + b[j];
  acc[i] = v > 0.0f ? v : (expf(v) - 1.0f);
}

// z = acc/s + b2 (1 head, mean over 1 head == identity), in place on d_out
__global__ void finalize2(float* __restrict__ out, const float* __restrict__ s,
                          const float* __restrict__ b, int N) {
  int i = blockIdx.x * blockDim.x + threadIdx.x;
  if (i >= N * 128) return;
  int n = i >> 7, c = i & 127;
  out[i] = out[i] / s[n] + b[c];
}

// ------------------------------- launcher ----------------------------------

extern "C" void kernel_launch(void* const* d_in, const int* in_sizes, int n_in,
                              void* d_out, int out_size, void* d_ws, size_t ws_size,
                              hipStream_t stream) {
  const float*     x    = (const float*)d_in[0];
  const long long* ei   = (const long long*)d_in[1];
  const float*     W1   = (const float*)d_in[2];
  const float*     aS1  = (const float*)d_in[3];
  const float*     aD1  = (const float*)d_in[4];
  const float*     b1   = (const float*)d_in[5];
  const float*     W2   = (const float*)d_in[6];
  const float*     aS2  = (const float*)d_in[7];
  const float*     aD2  = (const float*)d_in[8];
  const float*     b2   = (const float*)d_in[9];
  float* out = (float*)d_out;

  const int N = in_sizes[0] / 512;
  const int E = in_sizes[1] / 2;
  const long ET = (long)E + N;

  // workspace carve-out (256B aligned slabs)
  size_t off = 0;
  auto take = [&](size_t bytes) -> void* {
    void* p = (char*)d_ws + off;
    off = (off + bytes + 255) & ~(size_t)255;
    return p;
  };
  _Float16* w1T = (_Float16*)take((size_t)64 * 512 * sizeof(_Float16));
  _Float16* w2T = (_Float16*)take((size_t)128 * 64 * sizeof(_Float16));
  float* h1   = (float*)take((size_t)N * 64 * sizeof(float));
  float* as1  = (float*)take((size_t)N * 8 * sizeof(float));
  float* ad1  = (float*)take((size_t)N * 8 * sizeof(float));
  float* s1   = (float*)take((size_t)N * 8 * sizeof(float));
  float* acc1 = (float*)take((size_t)N * 64 * sizeof(float));
  float* h2   = (float*)take((size_t)N * 128 * sizeof(float));
  float* as2  = (float*)take((size_t)N * sizeof(float));
  float* ad2  = (float*)take((size_t)N * sizeof(float));
  float* s2   = (float*)take((size_t)N * sizeof(float));

  const int TB = 256;
  auto cdiv = [](long a, long b) { return (int)((a + b - 1) / b); };

  // zero accumulators (every call: deterministic for graph replay)
  fill_zero<<<cdiv((long)N * 8,   TB), TB, 0, stream>>>(s1,   (long)N * 8);
  fill_zero<<<cdiv((long)N * 64,  TB), TB, 0, stream>>>(acc1, (long)N * 64);
  fill_zero<<<cdiv((long)N,       TB), TB, 0, stream>>>(s2,   (long)N);
  fill_zero<<<cdiv((long)N * 128, TB), TB, 0, stream>>>(out,  (long)N * 128);

  // weight transpose+convert to f16 [n][k]
  convertT<<<cdiv(512 * 64, TB), TB, 0, stream>>>(W1, w1T, 512, 64);
  convertT<<<cdiv(64 * 128, TB), TB, 0, stream>>>(W2, w2T, 64, 128);

  // ---- layer 1 ----
  gemm_wmma<64, 64, 512><<<cdiv(N, 64), TB, 0, stream>>>(x, w1T, h1, N);
  scores_k<<<cdiv((long)N * 8, TB), TB, 0, stream>>>(h1, aS1, aD1, as1, ad1, N, 8, 8);
  edge_sum<8><<<cdiv(ET, TB), TB, 0, stream>>>(ei, E, N, as1, ad1, s1);
  edge_agg<8, 8><<<cdiv(ET, TB), TB, 0, stream>>>(ei, E, N, as1, ad1, h1, acc1);
  finalize1<<<cdiv((long)N * 64, TB), TB, 0, stream>>>(acc1, s1, b1, N);

  // ---- layer 2 ----
  gemm_wmma<32, 128, 64><<<cdiv(N, 32), TB, 0, stream>>>(acc1, w2T, h2, N);
  scores_k<<<cdiv((long)N, TB), TB, 0, stream>>>(h2, aS2, aD2, as2, ad2, N, 1, 128);
  edge_sum<1><<<cdiv(ET, TB), TB, 0, stream>>>(ei, E, N, as2, ad2, s2);
  edge_agg<1, 128><<<cdiv(ET, TB), TB, 0, stream>>>(ei, E, N, as2, ad2, h2, out);
  finalize2<<<cdiv((long)N * 128, TB), TB, 0, stream>>>(out, s2, b2, N);

  (void)n_in; (void)out_size; (void)ws_size;
}
